// Transformer_11304353923780
// MI455X (gfx1250) — matrix operand order repaired and checked by reading, compiled, not run on hardware
//
#include <hip/hip_runtime.h>
#include <hip/hip_bf16.h>

typedef __attribute__((ext_vector_type(16))) _Float16 v16h;
typedef __attribute__((ext_vector_type(8)))  _Float16 v8h;
typedef __attribute__((ext_vector_type(8)))  float    v8f;
typedef __attribute__((ext_vector_type(4)))  unsigned int u32x4;
typedef __attribute__((ext_vector_type(8)))  int          i32x8;
typedef __attribute__((ext_vector_type(4)))  int          i32x4;

#define WMMA_F16(a, b, c) \
  __builtin_amdgcn_wmma_f32_16x16x32_f16(false, (a), false, (b), (short)0, (c), false, false)

// Tensor Data Mover path (gfx1250). This toolchain exposes the 6-arg form:
//   (uint32x4 g0, int32x8 g1, int32x4 g2, int32x4 g3, int32x8, int cpol)
#if __has_builtin(__builtin_amdgcn_tensor_load_to_lds) && __has_builtin(__builtin_amdgcn_s_wait_tensorcnt)
#define USE_TDM 1
#else
#define USE_TDM 0
#endif

// ---------------------------------------------------------------------------
// Fragment loaders (wave32; A/B use this file's contiguous k order; C per CDNA5 ISA 7.12.2)
// A (16x32 f16): lane L -> row = L%16; halves 0-15: K = (L/16)*16 + h (contiguous)
// B (32x16 f16): lane L -> col = L%16; halves 0-15: K = (L/16)*16 + h (contiguous)
// C (16x16 f32): lane L, float r -> row = (L/16)*8 + r, col = L%16
// ---------------------------------------------------------------------------
static __device__ __forceinline__ v16h pack16(v8h lo, v8h hi) {
  v16h r;
#pragma unroll
  for (int i = 0; i < 8; i++) { r[i] = lo[i]; r[8 + i] = hi[i]; }
  return r;
}

static __device__ __forceinline__ v16h load_a_frag(const _Float16* __restrict__ A,
                                                   int lda, int row_base, int k_base,
                                                   int lane) {
  const int row = row_base + (lane & 15);
  const int g = lane >> 4;
  const _Float16* p = A + (size_t)row * lda + k_base + g * 16;
  return pack16(*(const v8h*)(p), *(const v8h*)(p + 8));
}

// Bt stored "N-major": Bt[n * ldb + k] == B[k][n] -> contiguous 32B per lane
static __device__ __forceinline__ v16h load_b_frag(const _Float16* __restrict__ Bt,
                                                   int ldb, int n_base, int k_base,
                                                   int lane) {
  const int col = n_base + (lane & 15);
  const int g = lane >> 4;
  const _Float16* p = Bt + (size_t)col * ldb + k_base + g * 16;
  return *(const v16h*)p;
}

static __device__ __forceinline__ float gelu_f(float x) {
  return 0.5f * x * (1.0f + erff(x * 0.7071067811865475f));
}

#if USE_TDM
// Issue a TDM load of a 32-row x 64-half tile (row stride 512 halves) to LDS.
// D# per ISA 8.3/8.4: group0 = count|lds_addr|global_addr|type=2;
// group1 = data_size=2B, tensor_dim0=512, tensor_dim1=8192, tile 64x32,
//          tensor_dim0_stride=512.
static __device__ __forceinline__ void tdm_load_v32x64(unsigned lds_off,
                                                       const _Float16* gsrc) {
  const unsigned long long ga = (unsigned long long)(size_t)gsrc;
  u32x4 g0 = {1u,
              lds_off,
              (unsigned)(ga & 0xFFFFFFFFu),
              ((unsigned)((ga >> 32) & 0x01FFFFFFu)) | 0x80000000u};
  i32x8 g1 = {(int)0x00010000u,   // [17:16]=1 (2-byte elements)
              (int)(512u << 16),  // tensor_dim0 low16 @ bits 63:48
              (int)0x20000000u,   // dim0 hi16=0 | tensor_dim1 low16=8192
              (int)(64u << 16),   // dim1 hi16=0 | tile_dim0=64
              (int)32,            // tile_dim1=32, tile_dim2=0
              (int)512,           // tensor_dim0_stride low32
              0, 0};
  i32x4 z4 = {0, 0, 0, 0};
  i32x8 z8 = {0, 0, 0, 0, 0, 0, 0, 0};
  __builtin_amdgcn_tensor_load_to_lds(g0, g1, z4, z4, z8, 0);
}
#endif

// ---------------------------------------------------------------------------
// Weight cast + transpose: W[l][k][n] f32 -> Wt[(l*6+w)][n][k] f16
// ---------------------------------------------------------------------------
__global__ __launch_bounds__(256) void castT_kernel(const float* __restrict__ Wq,
                                                    const float* __restrict__ Wk,
                                                    const float* __restrict__ Wv,
                                                    const float* __restrict__ Wo,
                                                    const float* __restrict__ W1,
                                                    const float* __restrict__ W2,
                                                    _Float16* __restrict__ out) {
  const size_t i = (size_t)blockIdx.x * 256 + threadIdx.x;  // over 24*512*512
  const int mat = (int)(i >> 18);
  const int rem = (int)(i & 262143);
  const int n = rem >> 9;
  const int k = rem & 511;
  const int l = mat / 6, w = mat % 6;
  const float* src;
  switch (w) {
    case 0: src = Wq; break;
    case 1: src = Wk; break;
    case 2: src = Wv; break;
    case 3: src = Wo; break;
    case 4: src = W1; break;
    default: src = W2; break;
  }
  out[i] = (_Float16)src[(size_t)l * 262144 + (size_t)k * 512 + n];
}

// ---------------------------------------------------------------------------
// LayerNorm over D=512: one wave per token, f32 in -> f16 out
// ---------------------------------------------------------------------------
__global__ __launch_bounds__(256) void ln_kernel(const float* __restrict__ X,
                                                 const float* __restrict__ G,
                                                 const float* __restrict__ Bv,
                                                 _Float16* __restrict__ Y) {
  const int token = blockIdx.x * 8 + (threadIdx.x >> 5);
  const int lane = threadIdx.x & 31;
  const float* row = X + (size_t)token * 512;
  float v[16];
  float s = 0.f, s2 = 0.f;
#pragma unroll
  for (int i = 0; i < 4; i++) {
    float4 t = *(const float4*)(row + i * 128 + lane * 4);
    v[i * 4 + 0] = t.x; v[i * 4 + 1] = t.y; v[i * 4 + 2] = t.z; v[i * 4 + 3] = t.w;
  }
#pragma unroll
  for (int i = 0; i < 16; i++) { s += v[i]; s2 += v[i] * v[i]; }
#pragma unroll
  for (int m = 1; m < 32; m <<= 1) {
    s += __shfl_xor(s, m, 32);
    s2 += __shfl_xor(s2, m, 32);
  }
  const float mu = s * (1.0f / 512.0f);
  const float var = s2 * (1.0f / 512.0f) - mu * mu;
  const float rs = rsqrtf(var + 1e-5f);
#pragma unroll
  for (int i = 0; i < 4; i++) {
#pragma unroll
    for (int j = 0; j < 4; j++) {
      const int idx = i * 128 + lane * 4 + j;
      const float val = (v[i * 4 + j] - mu) * rs * G[idx] + Bv[idx];
      Y[(size_t)token * 512 + idx] = (_Float16)val;
    }
  }
}

// ---------------------------------------------------------------------------
// GEMM: C[8192,512] = A[8192,512] @ W[512,512]  (W pre-transposed f16)
// Each wave: 32x64 output tile, K loop step 32 -> 8 WMMA per 12 b128 loads.
// EPI: 0 = plain -> f16 out; 1 = +bias +residual -> f32 out; 2 = +bias, GELU -> f16
// ---------------------------------------------------------------------------
template <int EPI>
__global__ __launch_bounds__(256) void gemm_kernel(const _Float16* __restrict__ A,
                                                   const _Float16* __restrict__ Wt,
                                                   const float* __restrict__ bias,
                                                   const float* __restrict__ res,
                                                   _Float16* __restrict__ outH,
                                                   float* __restrict__ outF) {
  const int lane = threadIdx.x & 31;
  const int wid = blockIdx.x * 8 + (threadIdx.x >> 5);
  const int mt = wid & 255;   // 256 m-tiles of 32 rows
  const int nb = wid >> 8;    // 8 n-blocks of 64 cols
  const int c = lane & 15, g = lane >> 4;
  v8f acc[2][4] = {};
  for (int k0 = 0; k0 < 512; k0 += 32) {
    __builtin_prefetch(A + (size_t)(mt * 32 + c) * 512 + k0 + 64, 0, 0);
    v16h a0 = load_a_frag(A, 512, mt * 32, k0, lane);
    v16h a1 = load_a_frag(A, 512, mt * 32 + 16, k0, lane);
#pragma unroll
    for (int t = 0; t < 4; t++) {
      v16h b = load_b_frag(Wt, 512, nb * 64 + t * 16, k0, lane);
      acc[0][t] = WMMA_F16(a0, b, acc[0][t]);
      acc[1][t] = WMMA_F16(a1, b, acc[1][t]);
    }
  }
#pragma unroll
  for (int i = 0; i < 2; i++) {
#pragma unroll
    for (int t = 0; t < 4; t++) {
#pragma unroll
      for (int r = 0; r < 8; r++) {
        const int row = mt * 32 + i * 16 + g * 8 + r;
        const int col = nb * 64 + t * 16 + c;
        float v = acc[i][t][r];
        if (EPI >= 1) v += bias[col];
        const size_t off = (size_t)row * 512 + col;
        if (EPI == 1)       outF[off] = res[off] + v;
        else if (EPI == 2)  outH[off] = (_Float16)gelu_f(v);
        else                outH[off] = (_Float16)v;
      }
    }
  }
}

// ---------------------------------------------------------------------------
// Flash attention: one wave per (b, head, 16-query tile). Key loop step 32.
// scale = D^-0.5 = 512^-0.5 folded into the Q fragments (once, pre-loop).
// V chunks staged to LDS by the TDM with double buffering: each iteration
// waits on the chunk it consumes (s_wait_tensorcnt 0), then immediately
// issues the DMA for the next chunk into the other buffer, hiding the DMA
// behind the S / softmax / P@V work.
// ---------------------------------------------------------------------------
__global__ __launch_bounds__(128) void attn_kernel(const _Float16* __restrict__ Q,
                                                   const _Float16* __restrict__ Kb,
                                                   const _Float16* __restrict__ Vb,
                                                   _Float16* __restrict__ O) {
  __shared__ __align__(32) _Float16 pLds[4][16 * 32];
  __shared__ __align__(32) _Float16 vLds[4][2][32 * 64];
  const int lane = threadIdx.x & 31;
  const int wlocal = threadIdx.x >> 5;
  const int wid = blockIdx.x * 4 + wlocal;
  const int qt = wid & 127;
  const int hd = (wid >> 7) & 7;
  const int b = wid >> 10;
  const int c = lane & 15;
  const int g = lane >> 4;
  const int col0 = hd * 64;
  const int rowq = b * 2048 + qt * 16;
  const float scale = 0.04419417382415922f;  // 512^-0.5

  v16h qa0 = load_a_frag(Q, 512, rowq, col0, lane);
  v16h qa1 = load_a_frag(Q, 512, rowq, col0 + 32, lane);
  // Fold the softmax scale into Q once (f16 halves; one-time cost).
#pragma unroll
  for (int i = 0; i < 16; i++) {
    qa0[i] = (_Float16)((float)qa0[i] * scale);
    qa1[i] = (_Float16)((float)qa1[i] * scale);
  }

  float mi[8], li[8];
  v8f o_acc[4] = {};
#pragma unroll
  for (int r = 0; r < 8; r++) { mi[r] = -1e30f; li[r] = 0.f; }

  _Float16* pS = pLds[wlocal];
  const _Float16* vBase = Vb + (size_t)(b * 2048) * 512 + col0;

#if USE_TDM
  unsigned voff[2];
  voff[0] = (unsigned)(size_t)(__attribute__((address_space(3))) void*)vLds[wlocal][0];
  voff[1] = (unsigned)(size_t)(__attribute__((address_space(3))) void*)vLds[wlocal][1];
  // Prologue: stage chunk 0 into buffer 0.
  tdm_load_v32x64(voff[0], vBase);
#endif

  int ibuf = 0;
  for (int kb = 0; kb < 2048; kb += 32, ibuf ^= 1) {
    // ---- Acquire the staged V chunk; kick off the next DMA ----
#if USE_TDM
    __builtin_amdgcn_s_wait_tensorcnt(0);
    if (kb + 32 < 2048)
      tdm_load_v32x64(voff[ibuf ^ 1], vBase + (size_t)(kb + 32) * 512);
    const _Float16* vS = vLds[wlocal][ibuf];
#else
    const _Float16* vS = vLds[wlocal][0];
    {
      const _Float16* src = vBase + (size_t)(kb + lane) * 512;
#pragma unroll
      for (int j = 0; j < 4; j++)
        *(v16h*)(vLds[wlocal][0] + lane * 64 + j * 16) = *(const v16h*)(src + j * 16);
    }
#endif
    // ---- S = (Q*scale) @ K^T for two 16-key subtiles ----
    v8f s[2];
#pragma unroll
    for (int t = 0; t < 2; t++) {
      v16h kf0 = load_b_frag(Kb, 512, b * 2048 + kb + t * 16, col0, lane);
      v16h kf1 = load_b_frag(Kb, 512, b * 2048 + kb + t * 16, col0 + 32, lane);
      v8f z = {};
      z = WMMA_F16(qa1, kf1, z);
      s[t] = WMMA_F16(qa0, kf0, z);
    }
    // ---- Online softmax update (row = g*8 + r per C layout) ----
    float alpha[8];
#pragma unroll
    for (int r = 0; r < 8; r++) {
      const float s0 = s[0][r];
      const float s1 = s[1][r];
      float m = fmaxf(s0, s1);
#pragma unroll
      for (int msk = 1; msk < 16; msk <<= 1) m = fmaxf(m, __shfl_xor(m, msk, 32));
      const float mn = fmaxf(mi[r], m);
      alpha[r] = __expf(mi[r] - mn);
      const float p0 = __expf(s0 - mn);
      const float p1 = __expf(s1 - mn);
      float rsum = p0 + p1;
#pragma unroll
      for (int msk = 1; msk < 16; msk <<= 1) rsum += __shfl_xor(rsum, msk, 32);
      li[r] = li[r] * alpha[r] + rsum;
      mi[r] = mn;
      pS[(g * 8 + r) * 32 + c] = (_Float16)p0;
      pS[(g * 8 + r) * 32 + 16 + c] = (_Float16)p1;
    }
    // ---- P C-layout -> A-fragment: two contiguous 8-half runs per lane ----
    v16h pa;
    {
      const _Float16* p = pS + c * 32 + g * 16;
      pa = pack16(*(const v8h*)(p), *(const v8h*)(p + 8));
    }
    // ---- O = O*alpha + P @ V (B-fragments gathered from LDS) ----
#pragma unroll
    for (int t = 0; t < 4; t++) {
      v16h vf;
#pragma unroll
      for (int hh = 0; hh < 16; hh++)
        vf[hh] = vS[(g * 16 + hh) * 64 + t * 16 + c];
#pragma unroll
      for (int r = 0; r < 8; r++) o_acc[t][r] *= alpha[r];
      o_acc[t] = WMMA_F16(pa, vf, o_acc[t]);
    }
  }
  // ---- Normalize and write f16 output ----
#pragma unroll
  for (int t = 0; t < 4; t++) {
#pragma unroll
    for (int r = 0; r < 8; r++) {
      const int row = rowq + g * 8 + r;
      const float val = o_acc[t][r] / li[r];
      O[(size_t)row * 512 + col0 + t * 16 + c] = (_Float16)val;
    }
  }
}

// ---------------------------------------------------------------------------
// Host orchestration (stream-only, graph-capturable)
// ---------------------------------------------------------------------------
extern "C" void kernel_launch(void* const* d_in, const int* in_sizes, int n_in,
                              void* d_out, int out_size, void* d_ws, size_t ws_size,
                              hipStream_t stream) {
  (void)in_sizes; (void)n_in; (void)out_size; (void)ws_size;
  const float* x = (const float*)d_in[0];
  const float* Wq = (const float*)d_in[1];
  const float* Wk = (const float*)d_in[2];
  const float* Wv = (const float*)d_in[3];
  const float* Wo = (const float*)d_in[4];
  const float* bo = (const float*)d_in[5];
  const float* ln1g = (const float*)d_in[6];
  const float* ln1b = (const float*)d_in[7];
  const float* W1 = (const float*)d_in[8];
  const float* b1 = (const float*)d_in[9];
  const float* W2 = (const float*)d_in[10];
  const float* b2 = (const float*)d_in[11];
  const float* ln2g = (const float*)d_in[12];
  const float* ln2b = (const float*)d_in[13];
  float* out = (float*)d_out;

  const size_t MT = 8192;  // B*N tokens
  char* ws = (char*)d_ws;
  float* h = (float*)ws;          ws += MT * 512 * sizeof(float);
  _Float16* wt = (_Float16*)ws;   ws += (size_t)24 * 262144 * sizeof(_Float16);
  _Float16* ybuf = (_Float16*)ws; ws += MT * 512 * sizeof(_Float16);
  _Float16* qbuf = (_Float16*)ws; ws += MT * 512 * sizeof(_Float16);
  _Float16* kbuf = (_Float16*)ws; ws += MT * 512 * sizeof(_Float16);
  _Float16* vbuf = (_Float16*)ws; ws += MT * 512 * sizeof(_Float16);
  _Float16* obuf = (_Float16*)ws; ws += MT * 512 * sizeof(_Float16);
  _Float16* ubuf = (_Float16*)ws; ws += MT * 512 * sizeof(_Float16);

  (void)hipMemcpyAsync(h, x, MT * 512 * sizeof(float), hipMemcpyDeviceToDevice, stream);
  castT_kernel<<<24576, 256, 0, stream>>>(Wq, Wk, Wv, Wo, W1, W2, wt);

  for (int l = 0; l < 4; l++) {
    const _Float16* wq_t = wt + (size_t)(l * 6 + 0) * 262144;
    const _Float16* wk_t = wt + (size_t)(l * 6 + 1) * 262144;
    const _Float16* wv_t = wt + (size_t)(l * 6 + 2) * 262144;
    const _Float16* wo_t = wt + (size_t)(l * 6 + 3) * 262144;
    const _Float16* w1_t = wt + (size_t)(l * 6 + 4) * 262144;
    const _Float16* w2_t = wt + (size_t)(l * 6 + 5) * 262144;

    ln_kernel<<<1024, 256, 0, stream>>>(h, ln1g + l * 512, ln1b + l * 512, ybuf);
    gemm_kernel<0><<<256, 256, 0, stream>>>(ybuf, wq_t, nullptr, nullptr, qbuf, nullptr);
    gemm_kernel<0><<<256, 256, 0, stream>>>(ybuf, wk_t, nullptr, nullptr, kbuf, nullptr);
    gemm_kernel<0><<<256, 256, 0, stream>>>(ybuf, wv_t, nullptr, nullptr, vbuf, nullptr);
    attn_kernel<<<1024, 128, 0, stream>>>(qbuf, kbuf, vbuf, obuf);
    gemm_kernel<1><<<256, 256, 0, stream>>>(obuf, wo_t, bo + l * 512, h, nullptr, h);
    ln_kernel<<<1024, 256, 0, stream>>>(h, ln2g + l * 512, ln2b + l * 512, ybuf);
    gemm_kernel<2><<<256, 256, 0, stream>>>(ybuf, w1_t, b1 + l * 512, nullptr, ubuf, nullptr);
    float* dst = (l == 3) ? out : h;
    gemm_kernel<1><<<256, 256, 0, stream>>>(ubuf, w2_t, b2 + l * 512, h, nullptr, dst);
  }
}
